// Causal_Frequency_Mamba_45483703665118
// MI455X (gfx1250) — matrix operand order repaired and checked by reading, not compiled, not run on hardware
//
#include <hip/hip_runtime.h>
#include <math.h>

typedef __attribute__((ext_vector_type(16))) _Float16 v16h;
typedef __attribute__((ext_vector_type(8)))  _Float16 v8h;
typedef __attribute__((ext_vector_type(8)))  float    v8f;
typedef __attribute__((ext_vector_type(2)))  float    v2f;

#define HH 384
#define WWI 384
#define HW (384*384)
#define NB 2

#if defined(__gfx1250__) && __has_builtin(__builtin_amdgcn_wmma_f32_16x16x4_f32)
#define HAVE_WMMA_F32 1
#else
#define HAVE_WMMA_F32 0
#endif

// Use the gfx1250 async global->LDS data mover for cooperative staging.
#if defined(__gfx1250__)
#define HAVE_ASYNC_LDS 1
#else
#define HAVE_ASYNC_LDS 0
#endif

#if HAVE_ASYNC_LDS
__device__ __forceinline__ unsigned lds_addr32(const void* p) {
  return (unsigned)(unsigned long long)(__attribute__((address_space(3))) const void*)p;
}
__device__ __forceinline__ void async_g2l_b128(void* lds, const void* g) {
  unsigned l = lds_addr32(lds);
  asm volatile("global_load_async_to_lds_b128 %0, %1, off"
               :: "v"(l), "v"(g) : "memory");
}
__device__ __forceinline__ void async_wait0() {
  asm volatile("s_wait_asynccnt 0x0" ::: "memory");
}
#endif

// ---------------------------------------------------------------- DFT matrices
__global__ void k_gen_dft(float* Wc, float* Ws) {
  int idx = blockIdx.x * blockDim.x + threadIdx.x;
  if (idx >= HW) return;
  int k = idx / WWI, n = idx % WWI;
  int m = (int)(((long long)k * (long long)n) % WWI);     // angle reduction
  float th = 6.2831853071795864769f * ((float)m / (float)WWI);
  Wc[idx] = cosf(th);
  Ws[idx] = sinf(th);
}

// ---------------------------------------------------------------- f32 WMMA GEMM
// C = alpha * A(MxK) @ B(KxN) + beta * C ; row-major; M%16==0, N%64==0.
// One wave computes a 16x64 tile: A fragment reused across 4 WMMAs.
__global__ void k_gemm_f32(const float* __restrict__ A, long sA,
                           const float* __restrict__ B, long sB,
                           float* __restrict__ C, long sC,
                           int M, int N, int K, float alpha, float beta) {
  int lane = threadIdx.x & 31;
  int wave = threadIdx.x >> 5;
  int tile = blockIdx.x * (blockDim.x >> 5) + wave;
  int tilesN = N >> 6;
  if (tile >= (M >> 4) * tilesN) return;                  // wave-uniform
  int tm = tile / tilesN, tn = tile % tilesN;
  const float* Ab = A + (long)blockIdx.y * sA;
  const float* Bb = B + (long)blockIdx.y * sB;
  float*       Cb = C + (long)blockIdx.y * sC;
  int ml = lane & 15, sel = lane >> 4;
  int m0 = tm << 4, n0 = tn << 6;
#if HAVE_WMMA_F32
  v8f acc0 = {}, acc1 = {}, acc2 = {}, acc3 = {};
  for (int k0 = 0; k0 < K; k0 += 4) {
    const float* ap = Ab + (long)(m0 + ml) * K + k0 + 2 * sel;
    v2f a; a.x = ap[0]; a.y = ap[1];
    __builtin_prefetch(ap + 32, 0, 1);
    const float* bp = Bb + (long)(k0 + 2 * sel) * N + n0 + ml;
    v2f b0v, b1v, b2v, b3v;
    b0v.x = bp[0];  b0v.y = bp[N];
    b1v.x = bp[16]; b1v.y = bp[N + 16];
    b2v.x = bp[32]; b2v.y = bp[N + 32];
    b3v.x = bp[48]; b3v.y = bp[N + 48];
    acc0 = __builtin_amdgcn_wmma_f32_16x16x4_f32(false, a, false, b0v, (short)0, acc0, false, false);
    acc1 = __builtin_amdgcn_wmma_f32_16x16x4_f32(false, a, false, b1v, (short)0, acc1, false, false);
    acc2 = __builtin_amdgcn_wmma_f32_16x16x4_f32(false, a, false, b2v, (short)0, acc2, false, false);
    acc3 = __builtin_amdgcn_wmma_f32_16x16x4_f32(false, a, false, b3v, (short)0, acc3, false, false);
  }
  v8f accs[4] = {acc0, acc1, acc2, acc3};
  for (int j = 0; j < 4; ++j)
    for (int r = 0; r < 8; ++r) {
      long o = (long)(m0 + r + 8 * sel) * N + (n0 + 16 * j + ml);
      float v = alpha * accs[j][r];
      if (beta != 0.0f) v += beta * Cb[o];
      Cb[o] = v;
    }
#else
  for (int j = 0; j < 4; ++j) {
    float acc[8] = {0.f,0.f,0.f,0.f,0.f,0.f,0.f,0.f};
    for (int k0 = 0; k0 < K; ++k0) {
      float bv = Bb[(long)k0 * N + n0 + 16 * j + ml];
      for (int r = 0; r < 8; ++r)
        acc[r] += Ab[(long)(m0 + r + 8 * sel) * K + k0] * bv;
    }
    for (int r = 0; r < 8; ++r) {
      long o = (long)(m0 + r + 8 * sel) * N + (n0 + 16 * j + ml);
      float v = alpha * acc[r];
      if (beta != 0.0f) v += beta * Cb[o];
      Cb[o] = v;
    }
  }
#endif
}

// ---------------------------------------------------------------- conv1 (Cin=2) + PReLU -> NHWC f16
__global__ void k_conv1(const float* __restrict__ Zr, const float* __restrict__ Zi,
                        const float* __restrict__ w, const float* __restrict__ bias,
                        const float* __restrict__ prelu_a, _Float16* __restrict__ out) {
  int idx = blockIdx.x * blockDim.x + threadIdx.x;
  if (idx >= NB * HW * 32) return;
  int co = idx & 31;
  int p  = (idx >> 5) % HW;
  int b  = (idx >> 5) / HW;
  int y = p / WWI, x = p % WWI;
  float acc = bias[co];
  for (int i = 0; i < 2; ++i) {
    const float* src = (i ? Zi : Zr) + (long)b * HW;
    for (int ky = 0; ky < 3; ++ky) {
      int yy = y + ky - 1; if ((unsigned)yy >= (unsigned)HH) continue;
      for (int kx = 0; kx < 3; ++kx) {
        int xx = x + kx - 1; if ((unsigned)xx >= (unsigned)WWI) continue;
        acc += src[yy * WWI + xx] * w[((co * 2 + i) * 3 + ky) * 3 + kx];
      }
    }
  }
  float a = prelu_a[0];
  out[idx] = (_Float16)(acc >= 0.f ? acc : a * acc);
}

// ---------------------------------------------------------------- pack OIHW f32 -> [O][tap*Cin+ci] f16
__global__ void k_pack_w(const float* __restrict__ w, _Float16* __restrict__ wpk,
                         int Cout, int Cin, int KH, int KW) {
  int Ktot = Cin * KH * KW;
  int idx = blockIdx.x * blockDim.x + threadIdx.x;
  if (idx >= Cout * Ktot) return;
  int o = idx / Ktot, j = idx % Ktot;
  int t = j / Cin, ci = j % Cin;
  int ky = t / KW, kx = t % KW;
  wpk[idx] = (_Float16)w[((o * Cin + ci) * KH + ky) * KW + kx];
}

// ---------------------------------------------------------------- implicit-GEMM conv, 32->32 ch, f16 WMMA
// One wave: 16 pixels x 32 cout; K-chunk per WMMA = 32 channels of one tap.
// Packed weights staged once per workgroup into LDS via async global->LDS.
__global__ void k_conv_wmma(const _Float16* __restrict__ in, const _Float16* __restrict__ wpk,
                            const float* __restrict__ bias, _Float16* __restrict__ out,
                            int KH, int KW, int act) {
  extern __shared__ _Float16 wlds[];                      // 32*Ktot halves
  int Ktot = 32 * KH * KW;
  int nvec = (32 * Ktot) >> 3;                            // 16-byte chunks
  for (int i = threadIdx.x; i < nvec; i += blockDim.x) {
#if HAVE_ASYNC_LDS
    async_g2l_b128(wlds + (i << 3), wpk + (i << 3));
#else
    *(uint4*)(wlds + (i << 3)) = *(const uint4*)(wpk + (i << 3));
#endif
  }
#if HAVE_ASYNC_LDS
  async_wait0();
#endif
  __syncthreads();

  int lane = threadIdx.x & 31;
  int wave = threadIdx.x >> 5;
  int tile = blockIdx.x * (blockDim.x >> 5) + wave;
  int tilesPerImg = HW / 16;
  if (tile >= NB * tilesPerImg) return;                   // wave-uniform
  int b  = tile / tilesPerImg;
  int p0 = (tile % tilesPerImg) * 16;
  int ml = lane & 15, sel = lane >> 4;
  int p = p0 + ml;
  int y = p / WWI, x = p % WWI;
  int ph = KH / 2, pw = KW / 2;
  union U { v16h v; v8h h[2]; };
  v8f acc0 = {}, acc1 = {};
  const _Float16* inb = in + (long)b * HW * 32;
  for (int t = 0; t < KH * KW; ++t) {
    int yy = y + t / KW - ph, xx = x + t % KW - pw;
    U a;
    if ((unsigned)yy < (unsigned)HH && (unsigned)xx < (unsigned)WWI) {
      const _Float16* src = inb + ((long)(yy * WWI + xx)) * 32 + sel * 16;
      a.h[0] = *(const v8h*)(src);
      a.h[1] = *(const v8h*)(src + 8);
    } else {
      v8h z = {}; a.h[0] = z; a.h[1] = z;
    }
    const _Float16* w0 = wlds + (long)ml * Ktot + t * 32 + sel * 16;
    const _Float16* w1 = wlds + (long)(16 + ml) * Ktot + t * 32 + sel * 16;
    U b0, b1;
    b0.h[0] = *(const v8h*)(w0); b0.h[1] = *(const v8h*)(w0 + 8);
    b1.h[0] = *(const v8h*)(w1); b1.h[1] = *(const v8h*)(w1 + 8);
    acc0 = __builtin_amdgcn_wmma_f32_16x16x32_f16(false, a.v, false, b0.v, (short)0, acc0, false, false);
    acc1 = __builtin_amdgcn_wmma_f32_16x16x32_f16(false, a.v, false, b1.v, (short)0, acc1, false, false);
  }
  _Float16* outb = out + (long)b * HW * 32;
  for (int r = 0; r < 8; ++r) {
    int pm = p0 + r + 8 * sel;
    float v0 = acc0[r] + bias[ml];
    float v1 = acc1[r] + bias[16 + ml];
    if (act == 1) { v0 = fmaxf(v0, 0.f); v1 = fmaxf(v1, 0.f); }
    outb[(long)pm * 32 + ml]      = (_Float16)v0;
    outb[(long)pm * 32 + 16 + ml] = (_Float16)v1;
  }
}

// ---------------------------------------------------------------- pa3 1x1 conv + sigmoid
__global__ void k_pa3(const _Float16* __restrict__ a2, const float* __restrict__ w,
                      const float* __restrict__ bias, float* __restrict__ a3) {
  int idx = blockIdx.x * blockDim.x + threadIdx.x;
  if (idx >= NB * HW) return;
  const _Float16* src = a2 + (long)idx * 32;
  for (int o = 0; o < 3; ++o) {
    float acc = bias[o];
    for (int c = 0; c < 32; ++c) acc += (float)src[c] * w[o * 32 + c];
    a3[idx * 3 + o] = 1.f / (1.f + expf(-acc));
  }
}

// ---------------------------------------------------------------- c = clamp(h,0,1); 3 logistic steps
__global__ void k_cpoly(const _Float16* __restrict__ h, const float* __restrict__ a3,
                        _Float16* __restrict__ c) {
  int idx = blockIdx.x * blockDim.x + threadIdx.x;
  if (idx >= NB * HW * 32) return;
  int pi = idx >> 5;
  float v = (float)h[idx];
  v = fminf(fmaxf(v, 0.f), 1.f);
  for (int i = 0; i < 3; ++i) { float a = a3[pi * 3 + i]; v = v + a * v * (1.f - v); }
  c[idx] = (_Float16)v;
}

// ---------------------------------------------------------------- conv3 32->1
__global__ void k_conv3(const _Float16* __restrict__ c, const float* __restrict__ w,
                        const float* __restrict__ bias, float* __restrict__ h3) {
  int idx = blockIdx.x * blockDim.x + threadIdx.x;
  if (idx >= NB * HW) return;
  int b = idx / HW, p = idx % HW;
  int y = p / WWI, x = p % WWI;
  float acc = bias[0];
  for (int ky = 0; ky < 3; ++ky) {
    int yy = y + ky - 1; if ((unsigned)yy >= (unsigned)HH) continue;
    for (int kx = 0; kx < 3; ++kx) {
      int xx = x + kx - 1; if ((unsigned)xx >= (unsigned)WWI) continue;
      const _Float16* src = c + ((long)b * HW + (long)(yy * WWI + xx)) * 32;
      for (int ci = 0; ci < 32; ++ci) acc += (float)src[ci] * w[(ci * 3 + ky) * 3 + kx];
    }
  }
  h3[idx] = acc;
}

// ---------------------------------------------------------------- patch embed (48x48, stride 48)
__global__ void k_patch(const float* __restrict__ h3, const float* __restrict__ w,
                        const float* __restrict__ bias, float* __restrict__ tok) {
  int idx = blockIdx.x * blockDim.x + threadIdx.x;
  if (idx >= NB * 64 * 32) return;
  int e = idx & 31, l = (idx >> 5) & 63, b = idx >> 11;
  int gh = l >> 3, gw = l & 7;
  const float* src = h3 + (long)b * HW + (gh * 48) * WWI + gw * 48;
  const float* wp = w + e * 2304;
  float acc = bias[e];
  for (int i = 0; i < 48; ++i)
    for (int j = 0; j < 48; ++j) acc += src[i * WWI + j] * wp[i * 48 + j];
  tok[idx] = acc;
}

// ---------------------------------------------------------------- VSS / selective scan block
__device__ __forceinline__ int trl(int l) { return ((l & 7) << 3) | (l >> 3); }

__global__ void k_vss(const float* __restrict__ tok,
                      const float* ln1_g, const float* ln1_b, const float* in_w,
                      const float* dw_w, const float* dw_b, const float* xproj_w,
                      const float* dt_w, const float* dt_b, const float* A_log,
                      const float* Ds, const float* on_g, const float* on_b,
                      const float* out_w, float* __restrict__ tok2) {
  extern __shared__ float sm[];
  float* xn    = sm;                 // 64*32
  float* xz    = xn + 64 * 32;       // 64*128  ([l][j]; j<64 xm, j>=64 z)
  float* xflat = xz + 64 * 128;      // 64*64   ([d][l]); reused as yls after scan
  float* xdbl  = xflat + 64 * 64;    // 4*6*64  ([k][c][l])
  float* ycomb = xdbl + 4 * 6 * 64;  // 64*64   ([l][d])
  int tid = threadIdx.x;
  int b = blockIdx.x;
  const float* tb = tok + b * 64 * 32;

  if (tid < 64) {                                         // ln1 per token
    const float* tr = tb + tid * 32;
    float mu = 0.f; for (int c = 0; c < 32; ++c) mu += tr[c]; mu *= (1.f / 32.f);
    float var = 0.f; for (int c = 0; c < 32; ++c) { float d0 = tr[c] - mu; var += d0 * d0; }
    float inv = rsqrtf(var * (1.f / 32.f) + 1e-5f);
    for (int c = 0; c < 32; ++c) xn[tid * 32 + c] = (tr[c] - mu) * inv * ln1_g[c] + ln1_b[c];
  }
  __syncthreads();
  for (int o = tid; o < 64 * 128; o += blockDim.x) {      // in_proj
    int l = o >> 7, j = o & 127;
    float acc = 0.f;
    const float* xr = xn + l * 32; const float* wr = in_w + j * 32;
    for (int c = 0; c < 32; ++c) acc += xr[c] * wr[c];
    xz[o] = acc;
  }
  __syncthreads();
  for (int o = tid; o < 64 * 64; o += blockDim.x) {       // depthwise conv + silu
    int d0 = o >> 6, l = o & 63;
    int h = l >> 3, w = l & 7;
    float acc = dw_b[d0];
    for (int ky = 0; ky < 3; ++ky) { int yy = h + ky - 1; if ((unsigned)yy >= 8u) continue;
      for (int kx = 0; kx < 3; ++kx) { int xx = w + kx - 1; if ((unsigned)xx >= 8u) continue;
        acc += xz[(yy * 8 + xx) * 128 + d0] * dw_w[d0 * 9 + ky * 3 + kx]; } }
    xflat[d0 * 64 + l] = acc / (1.f + expf(-acc));
  }
  __syncthreads();
  for (int o = tid; o < 4 * 6 * 64; o += blockDim.x) {    // x_proj
    int k = o / (6 * 64), c = (o / 64) % 6, l = o & 63;
    int li = (k == 0) ? l : (k == 1) ? trl(l) : (k == 2) ? (63 - l) : trl(63 - l);
    float acc = 0.f;
    const float* wr = xproj_w + (k * 6 + c) * 64;
    for (int d0 = 0; d0 < 64; ++d0) acc += xflat[d0 * 64 + li] * wr[d0];
    xdbl[o] = acc;
  }
  __syncthreads();
  for (int o = tid; o < 64 * 64; o += blockDim.x) ycomb[o] = 0.f;
  __syncthreads();
  {                                                        // selective scan: tid -> (k,d)
    int k = tid >> 6, d0 = tid & 63;
    float A0 = -expf(A_log[(k * 64 + d0) * 2 + 0]);
    float A1 = -expf(A_log[(k * 64 + d0) * 2 + 1]);
    float w0 = dt_w[(k * 64 + d0) * 2 + 0], w1 = dt_w[(k * 64 + d0) * 2 + 1];
    float bdt = dt_b[k * 64 + d0], Dv = Ds[k * 64 + d0];
    const float* dts0 = xdbl + (k * 6 + 0) * 64;
    const float* dts1 = xdbl + (k * 6 + 1) * 64;
    const float* Bs0  = xdbl + (k * 6 + 2) * 64;
    const float* Bs1  = xdbl + (k * 6 + 3) * 64;
    const float* Cs0  = xdbl + (k * 6 + 4) * 64;
    const float* Cs1  = xdbl + (k * 6 + 5) * 64;
    float h0 = 0.f, h1 = 0.f;
    for (int l = 0; l < 64; ++l) {
      int li = (k == 0) ? l : (k == 1) ? trl(l) : (k == 2) ? (63 - l) : trl(63 - l);
      float u = xflat[d0 * 64 + li];
      float dv = dts0[l] * w0 + dts1[l] * w1 + bdt;
      dv = (dv > 20.f) ? dv : log1pf(expf(dv));
      float xbu = dv * u;
      h0 = expf(dv * A0) * h0 + xbu * Bs0[l];
      h1 = expf(dv * A1) * h1 + xbu * Bs1[l];
      float yv = h0 * Cs0[l] + h1 * Cs1[l] + Dv * u;
      int lt = (k == 0) ? l : (k == 1) ? trl(l) : (k == 2) ? (63 - l) : trl(63 - l);
      atomicAdd(&ycomb[lt * 64 + d0], yv);
    }
  }
  __syncthreads();
  float* yls = xflat;                                      // reuse
  if (tid < 64) {                                          // out-LN * silu(z)
    int l = tid;
    float mu = 0.f; for (int d0 = 0; d0 < 64; ++d0) mu += ycomb[l * 64 + d0]; mu *= (1.f / 64.f);
    float var = 0.f; for (int d0 = 0; d0 < 64; ++d0) { float q = ycomb[l * 64 + d0] - mu; var += q * q; }
    float inv = rsqrtf(var * (1.f / 64.f) + 1e-5f);
    for (int d0 = 0; d0 < 64; ++d0) {
      float yn = (ycomb[l * 64 + d0] - mu) * inv * on_g[d0] + on_b[d0];
      float zv = xz[l * 128 + 64 + d0];
      yls[l * 64 + d0] = yn * (zv / (1.f + expf(-zv)));
    }
  }
  __syncthreads();
  for (int o = tid; o < 64 * 32; o += blockDim.x) {        // out proj + residual
    int l = o >> 5, e = o & 31;
    float acc = 0.f;
    const float* wr = out_w + e * 64;
    for (int d0 = 0; d0 < 64; ++d0) acc += yls[l * 64 + d0] * wr[d0];
    tok2[b * 64 * 32 + o] = tb[o] + acc;
  }
}

// ---------------------------------------------------------------- unembed + residual with x_f
__global__ void k_unemb(const float* __restrict__ tok2, const float* __restrict__ w,
                        const float* __restrict__ bias, const float* __restrict__ Zr,
                        const float* __restrict__ Zi, float* __restrict__ Ur,
                        float* __restrict__ Ui) {
  int idx = blockIdx.x * blockDim.x + threadIdx.x;
  if (idx >= NB * HW) return;
  int b = idx / HW, p = idx % HW;
  int py = p / WWI, px = p % WWI;
  int gh = py / 48, ph = py % 48, gw = px / 48, pw = px % 48;
  int l = gh * 8 + gw, r = ph * 48 + pw;
  const float* t = tok2 + (b * 64 + l) * 32;
  const float* wr = w + r * 32;
  float acc = bias[r];
  for (int e = 0; e < 32; ++e) acc += t[e] * wr[e];
  Ur[idx] = acc + Zr[idx];
  Ui[idx] = acc + Zi[idx];
}

// ---------------------------------------------------------------- final: scale + residual
__global__ void k_final(const float* __restrict__ Zo, const float* __restrict__ x,
                        float* __restrict__ out) {
  int idx = blockIdx.x * blockDim.x + threadIdx.x;
  if (idx >= NB * HW) return;
  out[idx] = Zo[idx] * (1.f / (float)HW) + x[idx];
}

// ================================================================ launcher
extern "C" void kernel_launch(void* const* d_in, const int* in_sizes, int n_in,
                              void* d_out, int out_size, void* d_ws, size_t ws_size,
                              hipStream_t stream) {
  (void)in_sizes; (void)n_in; (void)out_size; (void)ws_size;
  const float* x        = (const float*)d_in[0];
  const float* conv1_w  = (const float*)d_in[1];
  const float* conv1_b  = (const float*)d_in[2];
  const float* prelu_a  = (const float*)d_in[3];
  const float* conv2_w  = (const float*)d_in[4];
  const float* conv2_b  = (const float*)d_in[5];
  const float* pa1_w    = (const float*)d_in[6];
  const float* pa1_b    = (const float*)d_in[7];
  const float* pa2_w    = (const float*)d_in[8];
  const float* pa2_b    = (const float*)d_in[9];
  const float* pa3_w    = (const float*)d_in[10];
  const float* pa3_b    = (const float*)d_in[11];
  const float* conv3_w  = (const float*)d_in[12];
  const float* conv3_b  = (const float*)d_in[13];
  const float* patch_w  = (const float*)d_in[14];
  const float* patch_b  = (const float*)d_in[15];
  const float* ln1_g    = (const float*)d_in[16];
  const float* ln1_b    = (const float*)d_in[17];
  const float* in_proj_w= (const float*)d_in[18];
  const float* dw_w     = (const float*)d_in[19];
  const float* dw_b     = (const float*)d_in[20];
  const float* xproj_w  = (const float*)d_in[21];
  const float* dt_w     = (const float*)d_in[22];
  const float* dt_b     = (const float*)d_in[23];
  const float* A_log    = (const float*)d_in[24];
  const float* Ds       = (const float*)d_in[25];
  const float* on_g     = (const float*)d_in[26];
  const float* on_b     = (const float*)d_in[27];
  const float* out_w    = (const float*)d_in[28];
  const float* unemb_w  = (const float*)d_in[29];
  const float* unemb_b  = (const float*)d_in[30];

  char* wsb = (char*)d_ws;
  size_t off = 0;
  auto alloc = [&](size_t bytes) -> char* {
    char* p = wsb + off; off += (bytes + 255) & ~(size_t)255; return p;
  };
  float* Wc  = (float*)alloc((size_t)HW * 4);
  float* Wsm = (float*)alloc((size_t)HW * 4);
  float* Yr  = (float*)alloc((size_t)NB * HW * 4);
  float* Yi  = (float*)alloc((size_t)NB * HW * 4);
  float* Zr  = (float*)alloc((size_t)NB * HW * 4);
  float* Zi  = (float*)alloc((size_t)NB * HW * 4);
  float* Ur  = (float*)alloc((size_t)NB * HW * 4);
  float* Ui  = (float*)alloc((size_t)NB * HW * 4);
  float* Tr  = (float*)alloc((size_t)NB * HW * 4);
  float* Ti  = (float*)alloc((size_t)NB * HW * 4);
  float* Zo  = (float*)alloc((size_t)NB * HW * 4);
  _Float16* X0 = (_Float16*)alloc((size_t)NB * HW * 32 * 2);
  _Float16* X1 = (_Float16*)alloc((size_t)NB * HW * 32 * 2);
  _Float16* X2 = (_Float16*)alloc((size_t)NB * HW * 32 * 2);
  float* A3  = (float*)alloc((size_t)NB * HW * 3 * 4);
  float* H3  = (float*)alloc((size_t)NB * HW * 4);
  float* TOK = (float*)alloc((size_t)NB * 64 * 32 * 4);
  float* TOK2= (float*)alloc((size_t)NB * 64 * 32 * 4);
  _Float16* WPK2 = (_Float16*)alloc((size_t)32 * 288 * 2);
  _Float16* WPKA = (_Float16*)alloc((size_t)32 * 800 * 2);
  _Float16* WPKB = (_Float16*)alloc((size_t)32 * 288 * 2);

  auto gemm = [&](const float* A, long sA, const float* B, long sB,
                  float* C, long sC, float al, float be) {
    // 24 m-tiles x 6 n64-tiles = 144 wave-tiles; 4 waves/block -> 36 blocks.
    k_gemm_f32<<<dim3(36, NB), dim3(128), 0, stream>>>(A, sA, B, sB, C, sC,
                                                       384, 384, 384, al, be);
  };
  const int T256 = 256;
  #define GRD(n) dim3(((n) + T256 - 1) / T256)

  // -- forward FFT (DFT-as-GEMM): x_f = concat(Zr, Zi)
  k_gen_dft<<<GRD(HW), T256, 0, stream>>>(Wc, Wsm);
  gemm(x,   HW, Wc,  0, Yr, HW,  1.f, 0.f);   // Yr = x @ Wc
  gemm(x,   HW, Wsm, 0, Yi, HW, -1.f, 0.f);   // Yi = -x @ Ws
  gemm(Wc,  0,  Yr, HW, Zr, HW,  1.f, 0.f);   // Zr = Wc@Yr + Ws@Yi
  gemm(Wsm, 0,  Yi, HW, Zr, HW,  1.f, 1.f);
  gemm(Wc,  0,  Yi, HW, Zi, HW,  1.f, 0.f);   // Zi = Wc@Yi - Ws@Yr
  gemm(Wsm, 0,  Yr, HW, Zi, HW, -1.f, 1.f);

  // -- conv stack
  k_conv1<<<GRD(NB * HW * 32), T256, 0, stream>>>(Zr, Zi, conv1_w, conv1_b, prelu_a, X0);
  k_pack_w<<<GRD(32 * 288), T256, 0, stream>>>(conv2_w, WPK2, 32, 32, 3, 3);
  k_pack_w<<<GRD(32 * 800), T256, 0, stream>>>(pa1_w,   WPKA, 32, 32, 5, 5);
  k_pack_w<<<GRD(32 * 288), T256, 0, stream>>>(pa2_w,   WPKB, 32, 32, 3, 3);
  const int convBlocks = (NB * (HW / 16)) / 4;
  const size_t smem3 = (size_t)32 * 288 * 2;  // 18 KB
  const size_t smem5 = (size_t)32 * 800 * 2;  // 50 KB
  k_conv_wmma<<<convBlocks, 128, smem3, stream>>>(X0, WPK2, conv2_b, X1, 3, 3, 0); // h
  k_conv_wmma<<<convBlocks, 128, smem5, stream>>>(X1, WPKA, pa1_b,  X2, 5, 5, 1);  // a1
  k_conv_wmma<<<convBlocks, 128, smem3, stream>>>(X2, WPKB, pa2_b,  X0, 3, 3, 1);  // a2
  k_pa3<<<GRD(NB * HW), T256, 0, stream>>>(X0, pa3_w, pa3_b, A3);
  k_cpoly<<<GRD(NB * HW * 32), T256, 0, stream>>>(X1, A3, X2);                     // c
  k_conv3<<<GRD(NB * HW), T256, 0, stream>>>(X2, conv3_w, conv3_b, H3);

  // -- tokens, VSS block, unembed
  k_patch<<<GRD(NB * 64 * 32), T256, 0, stream>>>(H3, patch_w, patch_b, TOK);
  const size_t vss_smem = (size_t)(64*32 + 64*128 + 64*64 + 4*6*64 + 64*64) * 4;
  k_vss<<<NB, 256, vss_smem, stream>>>(TOK, ln1_g, ln1_b, in_proj_w, dw_w, dw_b,
                                       xproj_w, dt_w, dt_b, A_log, Ds, on_g, on_b,
                                       out_w, TOK2);
  k_unemb<<<GRD(NB * HW), T256, 0, stream>>>(TOK2, unemb_w, unemb_b, Zr, Zi, Ur, Ui);

  // -- inverse FFT (real part) + residual
  gemm(Ur,  HW, Wc,  0, Tr, HW,  1.f, 0.f);   // Tr = Ur@Wc - Ui@Ws
  gemm(Ui,  HW, Wsm, 0, Tr, HW, -1.f, 1.f);
  gemm(Ur,  HW, Wsm, 0, Ti, HW,  1.f, 0.f);   // Ti = Ur@Ws + Ui@Wc
  gemm(Ui,  HW, Wc,  0, Ti, HW,  1.f, 1.f);
  gemm(Wc,  0,  Tr, HW, Zo, HW,  1.f, 0.f);   // Re = Wc@Tr - Ws@Ti
  gemm(Wsm, 0,  Ti, HW, Zo, HW, -1.f, 1.f);
  k_final<<<GRD(NB * HW), T256, 0, stream>>>(Zo, x, (float*)d_out);
  #undef GRD
}
